// DeepseekV4LearnedRouter_73813307949706
// MI455X (gfx1250) — compile-verified, matrix operand order checked by reading, not run on hardware
//
#include <hip/hip_runtime.h>
#include <hip/hip_bf16.h>
#include <stdint.h>

// DeepseekV4 learned router, fused gate-GEMM + sqrt-softplus top-8 for MI455X (gfx1250).
// hidden[16384,4096] f32 x weight[256,4096] f32 -> bf16 WMMA (f32 acc) -> per-token top-8.
// Double-buffered async-to-LDS fp32 staging (ASYNCcnt), one cooperative fp32->bf16
// conversion pass per K-step, bf16 fragments ds_load'ed straight into WMMA sources.

typedef __attribute__((ext_vector_type(16))) __bf16 v16bf;
typedef __attribute__((ext_vector_type(8)))  __bf16 v8bf;
typedef __attribute__((ext_vector_type(4)))  __bf16 v4bf;
typedef __attribute__((ext_vector_type(8)))  float  v8f;

#define N_TOK   16384
#define DIM     4096
#define NEXP    256
#define BM      64
#define BK      32
#define TOPK    8
#define SCALE_F 2.5f

// Dynamic LDS layout (bytes)
#define A_BYTES (BM * BK * 4)                    //  8192  (A fp32 stage)
#define B_BYTES (NEXP * BK * 4)                  // 32768  (B fp32 stage)
#define STAGE_BYTES (A_BYTES + B_BYTES)          // 40960
#define L_OFF (2 * STAGE_BYTES)                  // 81920
// bf16 buffer (20 KB) overlaps the logits region (logits live only after GEMM)
#define SMEM_BYTES (L_OFF + BM * NEXP * 4)       // 147456 (144 KB) -> 2 WGs/WGP

#define STAGE_FLOATS ((BM + NEXP) * BK)          // 10240 floats = 2560 quads

__device__ __forceinline__ void async_b128(uint32_t lds_off, const void* g) {
    // VDST = per-lane LDS byte address, VADDR = 64-bit global address, no SADDR.
    asm volatile("global_load_async_to_lds_b128 %0, %1, off"
                 :: "v"(lds_off), "v"((uint64_t)(uintptr_t)g)
                 : "memory");
}

__device__ __forceinline__ v16bf load_frag_bf(const __bf16* p) {
    // Two ds_load_b128 straight into the fragment register pair halves.
    v8bf lo = *(const v8bf*)p;          // K chunk (half*8 .. +7)
    v8bf hi = *(const v8bf*)(p + 16);   // K chunk (+16)
    return __builtin_shufflevector(lo, hi, 0,1,2,3,4,5,6,7,8,9,10,11,12,13,14,15);
}

__global__ void __launch_bounds__(256)
router_fused_kernel(const float* __restrict__ hidden,
                    const float* __restrict__ weight,
                    const float* __restrict__ ebias,
                    float* __restrict__ out)
{
    extern __shared__ char smem[];
    float*  sL  = (float*)(smem + L_OFF);
    __bf16* bfb = (__bf16*)(smem + L_OFF);               // overlaps sL (disjoint lifetime)
    const uint32_t lds_base = (uint32_t)(uintptr_t)smem; // low 32 bits = LDS byte addr

    const int t      = threadIdx.x;
    const int lane   = t & 31;
    const int w      = t >> 5;        // wave id 0..7
    const int wave_m = w & 3;         // M-tile 0..3
    const int wave_n = w >> 2;        // N-half 0..1 (8 N-tiles each)
    const int m_base = blockIdx.x * BM;

    const int half = lane >> 4;       // 16-bit operand layout: K-half select
    const int r15  = lane & 15;       // row within tile

    v8f acc[8];
#pragma unroll
    for (int i = 0; i < 8; ++i) acc[i] = (v8f){0.f,0.f,0.f,0.f,0.f,0.f,0.f,0.f};

    // -------- staging: 10 async b128 per thread per K-step (2 for A, 8 for B) --------
    auto stage = [&](int buf, int k0) {
        uint32_t s_off = lds_base + (uint32_t)buf * STAGE_BYTES;
#pragma unroll
        for (int i = 0; i < 2; ++i) {               // A: 64 rows x 8 x 16B segs
            int id = t + i * 256, row = id >> 3, seg = id & 7;
            const float* g = hidden + (size_t)(m_base + row) * DIM + k0 + seg * 4;
            async_b128(s_off + (uint32_t)(row * BK + seg * 4) * 4, g);
        }
#pragma unroll
        for (int i = 0; i < 8; ++i) {               // B: 256 rows x 8 x 16B segs
            int id = t + i * 256, row = id >> 3, seg = id & 7;
            const float* g = weight + (size_t)row * DIM + k0 + seg * 4;
            async_b128(s_off + A_BYTES + (uint32_t)(row * BK + seg * 4) * 4, g);
        }
    };

    // ---------------- Phase 1: gate GEMM, logits kept in LDS ----------------
    stage(0, 0);
    for (int i = 0; i < DIM / BK; ++i) {
        const int cur    = i & 1;
        const int k_next = (i + 1) * BK;
        if (k_next < DIM) {
            stage(cur ^ 1, k_next);
            // async loads complete in order: <=10 outstanding => current buffer landed
            asm volatile("s_wait_asynccnt 10" ::: "memory");
        } else {
            asm volatile("s_wait_asynccnt 0" ::: "memory");
        }
        __syncthreads();   // fp32 stage[cur] visible to all waves

        // Cooperative conversion pass: each element converted exactly once.
        // 2560 float4 quads over 256 threads -> 10 per thread.
        {
            const float4* sf = (const float4*)(smem + cur * STAGE_BYTES);
#pragma unroll
            for (int q = 0; q < 10; ++q) {
                int qi = t + q * 256;
                float4 v = sf[qi];
                v4bf c;
                c[0] = (__bf16)v.x; c[1] = (__bf16)v.y;
                c[2] = (__bf16)v.z; c[3] = (__bf16)v.w;
                *(v4bf*)(bfb + qi * 4) = c;     // ds_store_b64
            }
        }
        __syncthreads();   // bf16 buffer ready

        const __bf16* bfA = bfb;                 // [64][32] bf16
        const __bf16* bfB = bfb + BM * BK;       // [256][32] bf16

        // A fragment: lanes 0-15 = row M, K 0..7 & 16..23; lanes 16-31: K 8..15 & 24..31
        v16bf af = load_frag_bf(bfA + (wave_m * 16 + r15) * BK + half * 8);

        const __bf16* bBr = bfB + r15 * BK + half * 8;
#pragma unroll
        for (int tt = 0; tt < 8; ++tt) {
            v16bf b = load_frag_bf(bBr + (wave_n * 8 + tt) * 16 * BK);
            acc[tt] = __builtin_amdgcn_wmma_f32_16x16x32_bf16(
                false, af, false, b, (short)0, acc[tt], false, false);
        }
        __syncthreads();   // bf16 buffer consumed before next conversion pass
    }

    // Dump accumulators: VGPR r -> M=r (lanes 0-15) / M=r+8 (lanes 16-31), N=lane&15
#pragma unroll
    for (int tt = 0; tt < 8; ++tt) {
        int col   = (wave_n * 8 + tt) * 16 + r15;
        int rbase = wave_m * 16 + half * 8;
#pragma unroll
        for (int r = 0; r < 8; ++r)
            sL[(rbase + r) * NEXP + col] = acc[tt][r];
    }
    __syncthreads();

    // ---------------- Phase 2: top-8 routing, one wave per token ----------------
    float* probs = out;
    float* rmap  = out + (size_t)N_TOK * NEXP;

    float eb[8];                       // selection bias, lane-constant across tokens
#pragma unroll
    for (int j = 0; j < 8; ++j) eb[j] = ebias[lane * 8 + j];

#pragma unroll 1
    for (int i = 0; i < 8; ++i) {
        int tl  = w * 8 + i;            // local token 0..63
        int tok = m_base + tl;

        float4 x0 = *(const float4*)(sL + tl * NEXP + lane * 8);
        float4 x1 = *(const float4*)(sL + tl * NEXP + lane * 8 + 4);
        float xv[8] = {x0.x, x0.y, x0.z, x0.w, x1.x, x1.y, x1.z, x1.w};

        float s[8], sel[8];
#pragma unroll
        for (int j = 0; j < 8; ++j) {
            float x  = xv[j];
            float sp = fmaxf(x, 0.f) + log1pf(__expf(-fabsf(x)));  // softplus, stable
            float sc = sqrtf(sp);
            s[j]   = sc;
            sel[j] = sc + eb[j];        // bias for selection only
        }

        unsigned mask  = 0u;
        float    denom = 0.f;
        for (int it = 0; it < TOPK; ++it) {
            float bv = -__builtin_inff(), bs = 0.f;
            int   bi = 0x7fffffff;
#pragma unroll
            for (int j = 0; j < 8; ++j) {
                int e = lane * 8 + j;
                if (sel[j] > bv || (sel[j] == bv && e < bi)) { bv = sel[j]; bs = s[j]; bi = e; }
            }
#pragma unroll
            for (int off = 16; off >= 1; off >>= 1) {   // wave32 butterfly argmax
                float ov = __shfl_xor(bv, off, 32);
                float os = __shfl_xor(bs, off, 32);
                int   oi = __shfl_xor(bi, off, 32);
                if (ov > bv || (ov == bv && oi < bi)) { bv = ov; bs = os; bi = oi; }
            }
            denom += bs;                       // un-biased score of winner
            if ((bi >> 3) == lane) {           // owner removes winner
                mask |= 1u << (bi & 7);
                sel[bi & 7] = -__builtin_inff();
            }
        }

        denom = fmaxf(denom, 1e-12f);
        float invs = SCALE_F / denom;
        float pv[8], mv[8];
#pragma unroll
        for (int j = 0; j < 8; ++j) {
            int on = (mask >> j) & 1;
            pv[j] = on ? s[j] * invs : 0.f;
            mv[j] = on ? 1.f : 0.f;
        }
        size_t o = (size_t)tok * NEXP + lane * 8;
        *(float4*)(probs + o)     = make_float4(pv[0], pv[1], pv[2], pv[3]);
        *(float4*)(probs + o + 4) = make_float4(pv[4], pv[5], pv[6], pv[7]);
        *(float4*)(rmap + o)      = make_float4(mv[0], mv[1], mv[2], mv[3]);
        *(float4*)(rmap + o + 4)  = make_float4(mv[4], mv[5], mv[6], mv[7]);
    }
}

extern "C" void kernel_launch(void* const* d_in, const int* in_sizes, int n_in,
                              void* d_out, int out_size, void* d_ws, size_t ws_size,
                              hipStream_t stream) {
    (void)in_sizes; (void)n_in; (void)out_size; (void)d_ws; (void)ws_size;
    const float* hidden = (const float*)d_in[0];
    const float* weight = (const float*)d_in[1];
    const float* ebias  = (const float*)d_in[2];
    float* out = (float*)d_out;

    dim3 grid(N_TOK / BM);   // 256 workgroups
    dim3 block(256);         // 8 wave32 waves
    hipLaunchKernelGGL(router_fused_kernel, grid, block, SMEM_BYTES, stream,
                       hidden, weight, ebias, out);
}
